// DendriteLayer_52596169507166
// MI455X (gfx1250) — matrix operand order repaired and checked by reading, not compiled, not run on hardware
//
#include <hip/hip_runtime.h>

typedef __attribute__((ext_vector_type(16))) __bf16       bf16x16;
typedef __attribute__((ext_vector_type(8)))  float        f32x8;
typedef __attribute__((ext_vector_type(4)))  unsigned int u32x4;

#define IN_DIM  2048
#define OUT_DIM 2048
#define DPC     8
#define ND      (OUT_DIM * DPC)   // 16384
#define BATCH   4096

union FragU { bf16x16 v; u32x4 q[2]; };

// Load a 16x32 bf16 fragment's per-lane 2x16B chunks via SADDR-form b128 loads:
// base is a uniform pointer (SGPR pair), offsets are 32-bit per-lane (GVS mode).
__device__ __forceinline__ bf16x16 load_frag(const __bf16* __restrict__ base,
                                             int off0, int off1) {
    FragU f;
    f.q[0] = *reinterpret_cast<const u32x4*>(base + off0);
    f.q[1] = *reinterpret_cast<const u32x4*>(base + off1);
    return f.v;
}

// ---------------- prologue: fp32 -> (hi, lo) bf16 split ----------------
__global__ void dl_split_x(const float* __restrict__ x,
                           __bf16* __restrict__ hi, __bf16* __restrict__ lo, int n) {
    for (int i = blockIdx.x * blockDim.x + threadIdx.x; i < n; i += gridDim.x * blockDim.x) {
        float v = x[i];
        __bf16 h = (__bf16)v;
        hi[i] = h;
        lo[i] = (__bf16)(v - (float)h);
    }
}

__global__ void dl_split_w(const float* __restrict__ w, const float* __restrict__ m,
                           __bf16* __restrict__ hi, __bf16* __restrict__ lo, int n) {
    for (int i = blockIdx.x * blockDim.x + threadIdx.x; i < n; i += gridDim.x * blockDim.x) {
        float v = w[i] * m[i];
        __bf16 h = (__bf16)v;
        hi[i] = h;
        lo[i] = (__bf16)(v - (float)h);
    }
}

__global__ void dl_boost(const float* __restrict__ duty, float* __restrict__ boost, int n) {
    int i = blockIdx.x * blockDim.x + threadIdx.x;
    if (i < n) boost[i] = __expf((0.125f - duty[i]) * 2.0f);
}

// ---------------- fused GEMM (split-bf16 WMMA) + k-winners + block-diag output ----------------
// grid: (OUT_DIM/16 column-groups, BATCH/128 row-blocks), 256 threads = 8 wave32.
// Each wave: 16 batch rows x (16 columns x all 8 dendrites) -> 8 f32x8 accumulators,
// so the boosted argmax over dendrites is lane-local.
__global__ __launch_bounds__(256)
void dl_fused(const __bf16* __restrict__ xhi, const __bf16* __restrict__ xlo,
              const __bf16* __restrict__ whi, const __bf16* __restrict__ wlo,
              const float* __restrict__ boost, const float* __restrict__ b_in,
              const float* __restrict__ w_out, const float* __restrict__ b_out,
              float* __restrict__ out) {
    const int lane = threadIdx.x & 31;
    const int wave = threadIdx.x >> 5;
    const int cg   = blockIdx.x;               // column group of 16
    const int c0   = cg * 16;
    const int m0   = blockIdx.y * 128 + wave * 16;
    const int l15  = lane & 15;
    const int half = lane >> 4;

    // A (16x32 bf16, M x K): lane row = m0+l15; chunks at K = half*8 and half*8+16.
    // 32-bit element offsets off uniform bases -> SADDR (GVS) addressing, no 64-bit carries.
    const int aoff = (m0 + l15) * IN_DIM + half * 8;

    // B (32x16 bf16, K x N): lane col n = c0+l15 (row j = d*2048+n of W[ND,IN]);
    // chunks at K = half*8 and half*8+16 (same order as A).
    int woffd[DPC];
#pragma unroll
    for (int d = 0; d < DPC; ++d)
        woffd[d] = (c0 + l15) * IN_DIM + half * 8 + d * (OUT_DIM * IN_DIM);

    f32x8 acc[DPC] = {};

    for (int k = 0; k < IN_DIM; k += 32) {
        bf16x16 ah = load_frag(xhi, aoff + k, aoff + k + 16);
        bf16x16 al = load_frag(xlo, aoff + k, aoff + k + 16);
#pragma unroll
        for (int d = 0; d < DPC; ++d) {
            bf16x16 bh = load_frag(whi, woffd[d] + k, woffd[d] + k + 16);
            bf16x16 bl = load_frag(wlo, woffd[d] + k, woffd[d] + k + 16);
            // fp32 emulation: hi*hi + hi*lo + lo*hi (lo*lo dropped, ~2^-16 rel err,
            // keeps the winner-take-all argmax faithful to the fp32 reference)
            acc[d] = __builtin_amdgcn_wmma_f32_16x16x32_bf16(false, ah, false, bh, (short)0, acc[d], false, false);
            acc[d] = __builtin_amdgcn_wmma_f32_16x16x32_bf16(false, ah, false, bl, (short)0, acc[d], false, false);
            acc[d] = __builtin_amdgcn_wmma_f32_16x16x32_bf16(false, al, false, bh, (short)0, acc[d], false, false);
        }
    }

    // ---- epilogue: boosted argmax over dendrites (lane-local), block-diag w_out dot ----
    const int c      = c0 + l15;               // this lane's column (C layout: N = lane%16)
    const int chalf  = (lane >> 3) & 1;        // (c%16)/8
    const int cgrp2  = cg * 2 + chalf;         // c/8
    float bst[DPC], bin[DPC], wfac[DPC];
#pragma unroll
    for (int d = 0; d < DPC; ++d) {
        bst[d] = boost[d * OUT_DIM + c];
        bin[d] = b_in[d * OUT_DIM + c];
        int u  = d * 256 + cgrp2;              // unit fed by dendrite-slice d, this c-octet
        wfac[d] = w_out[u * ND + d * OUT_DIM + c]; // == w_out[u, u*8 + c%8]
    }
    const int ddl = lane & 7;
    const int u_l = ddl * 256 + cgrp2;
    const float bo = b_out[u_l];

#pragma unroll
    for (int r = 0; r < 8; ++r) {              // C layout: M = r + 8*half
        float av[DPC];
#pragma unroll
        for (int d = 0; d < DPC; ++d) av[d] = acc[d][r] + bin[d];
        int win = 0; float best = av[0] * bst[0];
#pragma unroll
        for (int d = 1; d < DPC; ++d) {
            float f = av[d] * bst[d];
            if (f > best) { best = f; win = d; }   // strict > == argmax first-index tie-break
        }
        float s[DPC];
#pragma unroll
        for (int d = 0; d < DPC; ++d) s[d] = (win == d) ? av[d] * wfac[d] : 0.0f;
        // sum over the 8 lanes of this column-octet (3-step xor reduce, all lanes get result)
#pragma unroll
        for (int msk = 1; msk < 8; msk <<= 1) {
#pragma unroll
            for (int d = 0; d < DPC; ++d) s[d] += __shfl_xor(s[d], msk, 32);
        }
        float res = s[0];
#pragma unroll
        for (int d = 1; d < DPC; ++d) res = (ddl == d) ? s[d] : res;  // select without spilling
        int brow = m0 + r + half * 8;
        out[(size_t)brow * OUT_DIM + u_l] = res + bo;
    }
}

extern "C" void kernel_launch(void* const* d_in, const int* in_sizes, int n_in,
                              void* d_out, int out_size, void* d_ws, size_t ws_size,
                              hipStream_t stream) {
    const float* x     = (const float*)d_in[0];
    const float* w_in  = (const float*)d_in[1];
    const float* b_in  = (const float*)d_in[2];
    const float* mask  = (const float*)d_in[3];
    const float* w_out = (const float*)d_in[4];
    const float* b_out = (const float*)d_in[5];
    const float* duty  = (const float*)d_in[6];

    char* ws = (char*)d_ws;                    // ~160 MB used
    __bf16* xhi = (__bf16*)(ws);                         // 16 MB
    __bf16* xlo = (__bf16*)(ws + (size_t)16777216);      // 16 MB
    __bf16* whi = (__bf16*)(ws + (size_t)33554432);      // 64 MB
    __bf16* wlo = (__bf16*)(ws + (size_t)100663296);     // 64 MB
    float*  bst = (float*) (ws + (size_t)167772160);     // 64 KB

    dl_split_x<<<4096, 256, 0, stream>>>(x, xhi, xlo, BATCH * IN_DIM);
    dl_split_w<<<8192, 256, 0, stream>>>(w_in, mask, whi, wlo, ND * IN_DIM);
    dl_boost  <<<(DPC * OUT_DIM + 255) / 256, 256, 0, stream>>>(duty, bst, DPC * OUT_DIM);

    dim3 grid(OUT_DIM / 16, BATCH / 128);      // (128 column-groups, 32 row-blocks)
    dl_fused<<<grid, 256, 0, stream>>>(xhi, xlo, whi, wlo, bst, b_in, w_out, b_out, (float*)d_out);
}
